// CombineGraph_78116865180316
// MI455X (gfx1250) — compile-verified, matrix operand order checked by reading, not run on hardware
//
#include <hip/hip_runtime.h>

#define B_DIM 512
#define N_DIM 64
#define D_DIM 128
#define LALPHA 0.2f
#define NEG_INF -9e15f

typedef __attribute__((ext_vector_type(16))) _Float16 v16h;
typedef __attribute__((ext_vector_type(8)))  _Float16 v8h;
typedef __attribute__((ext_vector_type(8)))  float    v8f;

// LDS strides (elements), padded so every row start stays 16B aligned and
// rows land on different bank phases.
#define HA_STR 136   // h row-major f16: 128 + 8 pad   (272 B / row)
#define HT_STR 72    // h transposed f16: 64 + 8 pad   (144 B / row)
#define AL_STR 65    // alpha logits f32: 64 + 1 pad
#define AH_STR 72    // alpha softmax f16: 64 + 8 pad  (144 B / row)

// Load one lane's 16-half WMMA fragment (A layout; also valid for the B
// operand when B's per-lane column equals a row of the staged array).
// 16-bit 16x32 layout: lane<16 -> K = kc*32 + {0..7, 16..23}
//                      lane>=16 -> K = kc*32 + {8..15, 24..31}
__device__ __forceinline__ v16h load_frag(const _Float16* rowBase, int kc, int hs) {
    const v8h lo = *(const v8h*)(rowBase + kc * 32 + hs);
    const v8h hi = *(const v8h*)(rowBase + kc * 32 + 16 + hs);
    return __builtin_shufflevector(lo, hi, 0,1,2,3,4,5,6,7,8,9,10,11,12,13,14,15);
}

__global__ __launch_bounds__(256)
void gce_local_agg_kernel(const int* __restrict__ inputs,
                          const int* __restrict__ adj,
                          const float* __restrict__ embedding,
                          const float* __restrict__ attn_a,
                          float* __restrict__ out) {
    __shared__ __align__(16) _Float16 hA[N_DIM * HA_STR];     // 17408 B
    __shared__ __align__(16) _Float16 hT[D_DIM * HT_STR];     // 18432 B
    __shared__ __align__(16) _Float16 aF[4 * HA_STR];         //  1088 B
    __shared__ __align__(16) float    alphaF[N_DIM * AL_STR]; // 16640 B
    __shared__ __align__(16) _Float16 alphaH[N_DIM * AH_STR]; //  9216 B

    const int b    = blockIdx.x;
    const int t    = threadIdx.x;
    const int lane = t & 31;
    const int wave = t >> 5;

    // ---- Phase 0: attn_a -> LDS f16 ----
    for (int i = t; i < 4 * D_DIM; i += 256) {
        const int k = i >> 7, d = i & 127;
        aF[k * HA_STR + d] = (_Float16)attn_a[i];
    }

    // ---- Phase 1: embedding gather, stage h (f16) row-major + transposed ----
    {
        const int n = t >> 2;                 // 4 threads per row
        const int dbase = (t & 3) * 32;       // 32 contiguous floats each
        const int node = inputs[b * N_DIM + n];
        const float* src = embedding + (long)node * D_DIM + dbase;
        #pragma unroll 8
        for (int d = 0; d < 32; ++d) {
            const float f = src[d];
            const _Float16 hh = (_Float16)f;
            hA[n * HA_STR + dbase + d] = hh;
            hT[(dbase + d) * HT_STR + n] = hh;
        }
    }
    __syncthreads();

    // ---- Phase 2: e_k = leakyrelu((h .* a_k) h^T), adj-masked select ----
    // Wave w owns one i-row of tiles (i0 = (w>>1)*16) and two adjacent j-tiles
    // (j0 = (w&1)*32 .. +16), so each scaled A-fragment feeds two WMMAs.
    {
        const int hs = (lane >= 16) ? 8 : 0;
        const int ln = lane & 15;
        const int i0 = (wave >> 1) * 16;
        const int j0 = (wave & 1) * 32;
        const _Float16* rowA  = &hA[(i0 + ln) * HA_STR];
        const _Float16* rowB0 = &hA[(j0 + ln) * HA_STR];
        const _Float16* rowB1 = &hA[(j0 + 16 + ln) * HA_STR];
        v8f acc[4][2] = {};
        #pragma unroll
        for (int kc = 0; kc < 4; ++kc) {       // K = 128 in chunks of 32
            const v16h ha  = load_frag(rowA,  kc, hs);
            const v16h hb0 = load_frag(rowB0, kc, hs);
            const v16h hb1 = load_frag(rowB1, kc, hs);
            #pragma unroll
            for (int k = 0; k < 4; ++k) {
                const v16h ak = load_frag(&aF[k * HA_STR], kc, hs);
                const v16h s  = ha * ak;       // v_pk_mul_f16, shared by 2 WMMAs
                acc[k][0] = __builtin_amdgcn_wmma_f32_16x16x32_f16(
                    false, s, false, hb0, (short)0, acc[k][0], false, false);
                acc[k][1] = __builtin_amdgcn_wmma_f32_16x16x32_f16(
                    false, s, false, hb1, (short)0, acc[k][1], false, false);
            }
        }
        // C/D layout: lane<16 -> (M=r, N=lane); lane>=16 -> (M=r+8, N=lane-16)
        const int Mbase = i0 + ((lane >= 16) ? 8 : 0);
        #pragma unroll
        for (int jt = 0; jt < 2; ++jt) {
            const int Nc = j0 + jt * 16 + ln;
            #pragma unroll
            for (int r = 0; r < 8; ++r) {
                float e0 = acc[0][jt][r]; e0 = e0 > 0.f ? e0 : LALPHA * e0;
                float e1 = acc[1][jt][r]; e1 = e1 > 0.f ? e1 : LALPHA * e1;
                float e2 = acc[2][jt][r]; e2 = e2 > 0.f ? e2 : LALPHA * e2;
                float e3 = acc[3][jt][r]; e3 = e3 > 0.f ? e3 : LALPHA * e3;
                const int av = adj[((long)b * N_DIM + (Mbase + r)) * N_DIM + Nc];
                float v = NEG_INF;
                if      (av == 1) v = e0;
                else if (av == 2) v = e1;
                else if (av == 3) v = e2;
                else if (av == 4) v = e3;
                alphaF[(Mbase + r) * AL_STR + Nc] = v;
            }
        }
    }
    __syncthreads();

    // ---- Phase 3: row softmax (64 rows x 64), write f16 alpha ----
    // Thread t (< 64) owns row t exclusively, so in-place exp staging is safe.
    if (t < N_DIM) {
        float* row = &alphaF[t * AL_STR];
        float m = row[0];
        for (int j = 1; j < N_DIM; ++j) m = fmaxf(m, row[j]);
        float s = 0.f;
        for (int j = 0; j < N_DIM; ++j) {
            const float ev = __expf(row[j] - m);
            s += ev;
            row[j] = ev;
        }
        const float inv = 1.f / s;
        for (int j = 0; j < N_DIM; ++j)
            alphaH[t * AH_STR + j] = (_Float16)(row[j] * inv);
    }
    __syncthreads();

    // ---- Phase 4: out = alpha @ h  (64x64 x 64x128), f32 out ----
    // Wave w owns d-tile d0 = w*16 and all four i-tiles, so each B-fragment
    // (a column slice of h, read from hT) feeds four WMMAs.
    {
        const int hs = (lane >= 16) ? 8 : 0;
        const int ln = lane & 15;
        const int d0 = wave * 16;
        const _Float16* colB = &hT[(d0 + ln) * HT_STR]; // hT[d][j] = h[j][d]
        v8f acc[4] = {};
        #pragma unroll
        for (int kc = 0; kc < 2; ++kc) {       // K = 64 (j) in chunks of 32
            const v16h bb = load_frag(colB, kc, hs);
            #pragma unroll
            for (int it = 0; it < 4; ++it) {
                const v16h a = load_frag(&alphaH[(it * 16 + ln) * AH_STR], kc, hs);
                acc[it] = __builtin_amdgcn_wmma_f32_16x16x32_f16(
                    false, a, false, bb, (short)0, acc[it], false, false);
            }
        }
        const int Nc = d0 + ln;
        #pragma unroll
        for (int it = 0; it < 4; ++it) {
            const int Mbase = it * 16 + ((lane >= 16) ? 8 : 0);
            #pragma unroll
            for (int r = 0; r < 8; ++r)
                out[((long)b * N_DIM + (Mbase + r)) * D_DIM + Nc] = acc[it][r];
        }
    }
}

extern "C" void kernel_launch(void* const* d_in, const int* in_sizes, int n_in,
                              void* d_out, int out_size, void* d_ws, size_t ws_size,
                              hipStream_t stream) {
    (void)in_sizes; (void)n_in; (void)d_ws; (void)ws_size; (void)out_size;
    const int*   inputs    = (const int*)d_in[0];    // [B,N]
    const int*   adj       = (const int*)d_in[1];    // [B,N,N]
    // d_in[2] = mask_item, d_in[3] = item : unused by the reference output
    const float* embedding = (const float*)d_in[4];  // [NUM_NODE, D]
    const float* attn_a    = (const float*)d_in[5];  // [4, D]
    float* out = (float*)d_out;                      // [B,N,D]

    dim3 grid(B_DIM);
    dim3 block(256);
    gce_local_agg_kernel<<<grid, block, 0, stream>>>(inputs, adj, embedding, attn_a, out);
}
